// CustomNet_14147622273328
// MI455X (gfx1250) — matrix operand order repaired and checked by reading, not compiled, not run on hardware
//
#include <hip/hip_runtime.h>
#include <hip/hip_bf16.h>

typedef _Float16 v8h  __attribute__((ext_vector_type(8)));
typedef _Float16 v16h __attribute__((ext_vector_type(16)));
typedef float    v8f  __attribute__((ext_vector_type(8)));

// ---------------------------------------------------------------------------
// Workspace layout (bytes):
//   w2t  : [64][288] fp16, K = (dy*3+dx)*32 + c            @ 0        (36,864 B)
//   fc1t : [128][9216] fp16, K' = (y*12+x)*64 + c          @ 36,864   (2,359,296 B)
//   h1   : conv1 out, NHWC fp16 [2048][26][26][32]         @ 2,396,160 (88,604,672 B)
//   pooled: NHWC fp16 [2048][12][12][64]                   @ 91,000,832 (37,748,736 B)
// ---------------------------------------------------------------------------
#define WS_OFF_W2T   0
#define WS_OFF_FC1T  36864
#define WS_OFF_H1    2396160
#define WS_OFF_POOL  91000832

// A fragment halves live at p[0..7] and p[16..23] (K-split across lane halves)
__device__ __forceinline__ v16h load_split16(const _Float16* __restrict__ p) {
    v8h lo = *(const v8h*)p;
    v8h hi = *(const v8h*)(p + 16);
    return __builtin_shufflevector(lo, hi, 0, 1, 2, 3, 4, 5, 6, 7,
                                   8, 9, 10, 11, 12, 13, 14, 15);
}
// B fragment: 16 contiguous halves
__device__ __forceinline__ v16h load_cont16(const _Float16* __restrict__ p) {
    v8h lo = *(const v8h*)p;
    v8h hi = *(const v8h*)(p + 8);
    return __builtin_shufflevector(lo, hi, 0, 1, 2, 3, 4, 5, 6, 7,
                                   8, 9, 10, 11, 12, 13, 14, 15);
}

// ---------------- prep: repack weights to fp16 in WMMA-friendly layouts ----
__global__ void prep_kernel(const float* __restrict__ w2,
                            const float* __restrict__ fc1_w,
                            _Float16* __restrict__ w2t,
                            _Float16* __restrict__ fc1t) {
    int id = blockIdx.x * 256 + threadIdx.x;
    if (id < 64 * 288) {
        int o = id / 288, k = id % 288;
        int t = k / 32, c = k % 32;
        int dy = t / 3, dx = t % 3;
        // w2 is OIHW [64][32][3][3]
        w2t[o * 288 + k] = (_Float16)w2[((o * 32 + c) * 3 + dy) * 3 + dx];
    } else if (id < 64 * 288 + 128 * 9216) {
        int id2 = id - 64 * 288;
        int n = id2 / 9216, k = id2 % 9216;      // k = (y*12+x)*64 + c  (my layout)
        int yx = k / 64, c = k % 64;
        // reference flatten order: kref = c*144 + (y*12+x)
        fc1t[(size_t)n * 9216 + k] = (_Float16)fc1_w[(size_t)n * 9216 + c * 144 + yx];
    }
}

// ---------------- conv1: direct fp32 conv, ReLU, fp16 NHWC output ----------
__global__ void conv1_kernel(const float* __restrict__ x,
                             const float* __restrict__ w1,
                             const float* __restrict__ b1,
                             _Float16* __restrict__ h1) {
    __shared__ float sw[288];
    __shared__ float sb[32];
    int t = threadIdx.x;
    for (int i = t; i < 288; i += 256) sw[i] = w1[i];
    if (t < 32) sb[t] = b1[t];
    __syncthreads();

    int id = blockIdx.x * 256 + t;                 // one thread = one (b,y,x)
    if (id >= 2048 * 676) return;
    int b = id / 676, r = id % 676;
    int y = r / 26, xx = r % 26;
    const float* xp = x + (size_t)b * 784 + y * 28 + xx;

    float v[9];
#pragma unroll
    for (int ky = 0; ky < 3; ky++)
#pragma unroll
        for (int kx = 0; kx < 3; kx++) v[ky * 3 + kx] = xp[ky * 28 + kx];

    v16h lo{}, hi{};
#pragma unroll
    for (int c = 0; c < 32; c++) {
        float a = sb[c];
#pragma unroll
        for (int i = 0; i < 9; i++) a = fmaf(sw[c * 9 + i], v[i], a);
        a = a > 0.0f ? a : 0.0f;
        if (c < 16) lo[c] = (_Float16)a; else hi[c - 16] = (_Float16)a;
    }
    _Float16* op = h1 + (size_t)id * 32;           // 64-byte aligned
    *(v16h*)op = lo;
    *(v16h*)(op + 16) = hi;
}

// ---------------- conv2 implicit-GEMM WMMA + fused bias/ReLU/maxpool -------
// wave = one 4x4 spatial tile of one image, all 64 output channels.
// M-row m (=lane&15) -> conv pixel (y0 + m/4, x0 + m%4).
// A fragment (16x32 f16): lanes 0-15 hold K 0-15, lanes 16-31 hold
// K 16-31 (two contiguous 16B loads from NHWC h1).
// B fragment (32x16 f16): lanes 0-15 hold K 0-15, lanes 16-31 K 16-31, N=lane&15.
__global__ void conv2_kernel(const _Float16* __restrict__ h1,
                             const _Float16* __restrict__ w2t,
                             const float* __restrict__ b2,
                             _Float16* __restrict__ pooled) {
    int lane = threadIdx.x & 31;
    int wave = threadIdx.x >> 5;
    int tile = blockIdx.x * 8 + wave;              // 73728 tiles = 2048*36
    int b = tile / 36, tt = tile % 36;
    int ty = tt / 6, tx = tt % 6;
    int m = lane & 15;
    int y = ty * 4 + (m >> 2);
    int xx = tx * 4 + (m & 3);
    int kloA = (lane < 16) ? 0 : 16;
    int kloB = (lane < 16) ? 0 : 16;
    int nn = lane & 15;

    const _Float16* bbase = w2t + (size_t)nn * 288 + kloB;
    v8f acc[4] = {v8f{}, v8f{}, v8f{}, v8f{}};

#pragma unroll
    for (int t = 0; t < 9; t++) {
        int dy = t / 3, dx = t % 3;
        const _Float16* ap =
            h1 + (size_t)((b * 26 + y + dy) * 26 + (xx + dx)) * 32 + kloA;
        v16h a = load_cont16(ap);
        v16h bb[4];
#pragma unroll
        for (int j = 0; j < 4; j++)
            bb[j] = load_cont16(bbase + (size_t)j * (16 * 288) + t * 32);
#pragma unroll
        for (int j = 0; j < 4; j++)
            acc[j] = __builtin_amdgcn_wmma_f32_16x16x32_f16(
                false, a, false, bb[j], (short)0, acc[j], false, false);
    }

    // D layout: lanes 0-15 hold rows 0-7 (VGPR r -> M=r), lanes 16-31 rows 8-15.
    // Pool output Y = lane>>4; pooled(Y,X) = max of VGPRs {0,1,4,5} (X=0) and
    // {2,3,6,7} (X=1) within the lane's half.
    int Y = lane >> 4;
    int py = ty * 2 + Y;
    size_t rowbase = ((size_t)b * 12 + py) * 12;
#pragma unroll
    for (int j = 0; j < 4; j++) {
        int n = 16 * j + nn;
        float bias = b2[n];
        float f[8];
#pragma unroll
        for (int r = 0; r < 8; r++) {
            float v = acc[j][r] + bias;
            f[r] = v > 0.0f ? v : 0.0f;
        }
        float p0 = fmaxf(fmaxf(f[0], f[1]), fmaxf(f[4], f[5]));
        float p1 = fmaxf(fmaxf(f[2], f[3]), fmaxf(f[6], f[7]));
        pooled[(rowbase + tx * 2 + 0) * 64 + n] = (_Float16)p0;
        pooled[(rowbase + tx * 2 + 1) * 64 + n] = (_Float16)p1;
    }
}

// ---------------- fc1 (WMMA, K split over 8 waves) + ReLU + fc2 ------------
// Block = 256 threads = 8 waves, one block per 16 batch rows.
// Each wave computes a 16x128 partial over 1/8 of K (36 K-steps of 32) with
// explicit double-buffering (next step's A + 8 B fragments load while the
// current 8 WMMAs execute). Partials reduced through LDS, then bias/ReLU/fc2.
__global__ void fc_kernel(const _Float16* __restrict__ pooled,
                          const _Float16* __restrict__ fc1t,
                          const float* __restrict__ fc1_b,
                          const float* __restrict__ fc2_w,
                          const float* __restrict__ fc2_b,
                          float* __restrict__ out) {
    __shared__ float part[8][16 * 128];            // 64 KB
    int lane = threadIdx.x & 31;
    int wave = threadIdx.x >> 5;
    int b0 = blockIdx.x * 16;
    int m = lane & 15;
    int kloA = (lane < 16) ? 0 : 16;
    int kloB = (lane < 16) ? 0 : 16;

    const _Float16* arow = pooled + (size_t)(b0 + m) * 9216 + kloA;
    const _Float16* bbase = fc1t + (size_t)m * 9216 + kloB;

    int kk0 = wave * 36;                           // this wave's K-step range

    v8f acc[8] = {v8f{}, v8f{}, v8f{}, v8f{}, v8f{}, v8f{}, v8f{}, v8f{}};
    v16h aCur = load_cont16(arow + kk0 * 32);
    v16h bCur[8];
#pragma unroll
    for (int j = 0; j < 8; j++)
        bCur[j] = load_cont16(bbase + (size_t)j * (16 * 9216) + kk0 * 32);

#pragma unroll 1
    for (int s = 0; s < 36; s++) {
        int nk = kk0 + ((s < 35) ? s + 1 : 0);     // last-iter load is a dummy
        v16h aN = load_cont16(arow + nk * 32);
        v16h bN[8];
#pragma unroll
        for (int j = 0; j < 8; j++)
            bN[j] = load_cont16(bbase + (size_t)j * (16 * 9216) + nk * 32);
#pragma unroll
        for (int j = 0; j < 8; j++)
            acc[j] = __builtin_amdgcn_wmma_f32_16x16x32_f16(
                false, aCur, false, bCur[j], (short)0, acc[j], false, false);
        aCur = aN;
#pragma unroll
        for (int j = 0; j < 8; j++) bCur[j] = bN[j];
    }

    // write per-wave partials: row = r + 8*(lane>>4), n = 16*j + m
    int rbase = 8 * (lane >> 4);
#pragma unroll
    for (int j = 0; j < 8; j++) {
        int n = 16 * j + m;
#pragma unroll
        for (int r = 0; r < 8; r++)
            part[wave][(rbase + r) * 128 + n] = acc[j][r];
    }
    __syncthreads();

    // reduce 8 partials, add bias, ReLU; result back into part[0]
#pragma unroll
    for (int i = 0; i < 8; i++) {
        int idx = threadIdx.x + 256 * i;           // 2048 outputs, 8 per thread
        float s = 0.0f;
#pragma unroll
        for (int w = 0; w < 8; w++) s += part[w][idx];
        s += fc1_b[idx & 127];
        part[0][idx] = s > 0.0f ? s : 0.0f;        // exclusive per-thread slot
    }
    __syncthreads();

    // fc2: 16 rows x 10 outputs = 160 dot products of length 128
    if (threadIdx.x < 160) {
        int row = threadIdx.x / 10, o = threadIdx.x % 10;
        float s = fc2_b[o];
#pragma unroll 4
        for (int k = 0; k < 128; k++)
            s = fmaf(part[0][row * 128 + k], fc2_w[o * 128 + k], s);
        out[(size_t)(b0 + row) * 10 + o] = s;
    }
}

// ---------------------------------------------------------------------------
extern "C" void kernel_launch(void* const* d_in, const int* in_sizes, int n_in,
                              void* d_out, int out_size, void* d_ws, size_t ws_size,
                              hipStream_t stream) {
    const float* x     = (const float*)d_in[0];   // [2048,1,28,28]
    const float* w1    = (const float*)d_in[1];   // [32,1,3,3]
    const float* b1    = (const float*)d_in[2];   // [32]
    const float* w2    = (const float*)d_in[3];   // [64,32,3,3]
    const float* b2    = (const float*)d_in[4];   // [64]
    const float* fc1_w = (const float*)d_in[5];   // [128,9216]
    const float* fc1_b = (const float*)d_in[6];   // [128]
    const float* fc2_w = (const float*)d_in[7];   // [10,128]
    const float* fc2_b = (const float*)d_in[8];   // [10]
    float* out = (float*)d_out;                   // [2048,10]

    char* ws = (char*)d_ws;
    _Float16* w2t    = (_Float16*)(ws + WS_OFF_W2T);
    _Float16* fc1t   = (_Float16*)(ws + WS_OFF_FC1T);
    _Float16* h1     = (_Float16*)(ws + WS_OFF_H1);
    _Float16* pooled = (_Float16*)(ws + WS_OFF_POOL);

    // 64*288 + 128*9216 = 1,198,080 elements -> 4680 blocks of 256
    prep_kernel<<<4680, 256, 0, stream>>>(w2, fc1_w, w2t, fc1t);
    // 2048*676 = 1,384,448 threads -> 5408 blocks of 256
    conv1_kernel<<<5408, 256, 0, stream>>>(x, w1, b1, h1);
    // 2048*36 tiles, 8 waves/block -> 9216 blocks of 256
    conv2_kernel<<<9216, 256, 0, stream>>>(h1, w2t, b2, pooled);
    // 2048/16 rows, K split over 8 waves -> 128 blocks of 256
    fc_kernel<<<128, 256, 0, stream>>>(pooled, fc1t, fc1_b, fc2_w, fc2_b, out);
}
